// InterFrameAttention_24283745091599
// MI455X (gfx1250) — matrix operand order repaired and checked by reading, not compiled, not run on hardware
//
#include <hip/hip_runtime.h>
#include <hip/hip_bf16.h>

typedef __attribute__((ext_vector_type(16))) __bf16 bf16x16;
typedef __attribute__((ext_vector_type(8)))  float  f32x8;

union B16Frag { uint4 q[2]; unsigned int u[8]; bf16x16 v; };

// Packed f32 -> bf16 conversion: one VALU op (v_cvt_pk_bf16_f32).
static __device__ __forceinline__ unsigned int pack2(float a, float b) {
    unsigned int r;
    asm("v_cvt_pk_bf16_f32 %0, %1, %2" : "=v"(r) : "v"(a), "v"(b));
    return r;
}
static __device__ __forceinline__ unsigned short f2bf(float f) {
    return (unsigned short)(pack2(f, f) & 0xFFFFu);
}
static __device__ __forceinline__ f32x8 zero8() {
    f32x8 z = {0.f,0.f,0.f,0.f,0.f,0.f,0.f,0.f};
    return z;
}
static __device__ __forceinline__ f32x8 wmma_bf16(bf16x16 a, bf16x16 b, f32x8 c) {
    return __builtin_amdgcn_wmma_f32_16x16x32_bf16(false, a, false, b, (short)0, c, false, false);
}

// A fragment (16x32, M x K) from row-major bf16 buffer. lane elem e:
//   M = lane&15, K = (e&7) + 8*(lane>>4) + 16*(e>>3)
static __device__ __forceinline__ bf16x16 load_frag_a(const unsigned short* p, int ld, int row0, int k0) {
    int lane = threadIdx.x & 31;
    const unsigned short* r = p + (size_t)(row0 + (lane & 15)) * ld + k0 + 8 * (lane >> 4);
    B16Frag f;
    f.q[0] = *(const uint4*)(r);
    f.q[1] = *(const uint4*)(r + 16);
    return f.v;
}
// B fragment (32x16, K x N) from N-major ("transposed") bf16 buffer. lane elem e:
//   N = lane&15, K = (e&7) + 8*(lane>>4) + 16*(e>>3)
static __device__ __forceinline__ bf16x16 load_frag_b(const unsigned short* p, int ld, int n0, int k0) {
    int lane = threadIdx.x & 31;
    const unsigned short* r = p + (size_t)(n0 + (lane & 15)) * ld + k0 + 8 * (lane >> 4);
    B16Frag f;
    f.q[0] = *(const uint4*)(r);
    f.q[1] = *(const uint4*)(r + 16);
    return f.v;
}

#define BATCH     256
#define DIM_      256
#define NHEADS    8
#define NWIN      64
#define MROWS     65536           // B*N
#define SCALE_F   0.17677669529663687f

// ---------------------------------------------------------------------------
// Kernel 1: Y_bf16[M,N] = Xf32[M,K] @ Wf32[K,N].  Block: 256 thr (8 waves),
// tile 64(M) x 256(N), K-step 32. Waves 2x4 over (M,N); each 32x64 (2x4 WMMA).
// ---------------------------------------------------------------------------
__global__ void __launch_bounds__(256)
gemm_f32_bf16(const float* __restrict__ X, const float* __restrict__ W,
              unsigned short* __restrict__ Y, int K, int N) {
    __shared__ __align__(16) unsigned short As[64 * 40];
    __shared__ __align__(16) unsigned short Bs[256 * 40];

    const int t    = threadIdx.x;
    const int wave = t >> 5;
    const int lane = t & 31;
    const int half = lane >> 4;
    const int wm   = wave >> 2;     // 0..1
    const int wn   = wave & 3;      // 0..3
    const int mb   = blockIdx.x * 64;
    const int nb   = blockIdx.y * 256;

    f32x8 acc[2][4];
    #pragma unroll
    for (int mt = 0; mt < 2; ++mt)
        #pragma unroll
        for (int nt = 0; nt < 4; ++nt) acc[mt][nt] = zero8();

    const int nk = K / 32;
    for (int ks = 0; ks < nk; ++ks) {
        __syncthreads();
        // stage A tile: 64 x 32 f32 -> bf16; thread: row = t>>2, 8 cols.
        {
            const int row = t >> 2, kl = (t & 3) * 8;
            const float* xs = X + (size_t)(mb + row) * K + ks * 32 + kl;
            float4 x0 = *(const float4*)(xs);
            float4 x1 = *(const float4*)(xs + 4);
            if (ks + 1 < nk) __builtin_prefetch(xs + 32, 0, 1);
            uint4 pa;
            pa.x = pack2(x0.x, x0.y); pa.y = pack2(x0.z, x0.w);
            pa.z = pack2(x1.x, x1.y); pa.w = pack2(x1.z, x1.w);
            *(uint4*)&As[row * 40 + kl] = pa;
        }
        // stage W slice transposed: Bs[n][k] = W[ks*32+k][nb+n], 32 x 256.
        // thread: k-pair = 2*(t>>4), 16 n cols; b32 stores of (k, k+1) pairs.
        {
            const int kp = (t >> 4) * 2;
            const int nl = (t & 15) * 16;
            const float* w0 = W + (size_t)(ks * 32 + kp) * N + nb + nl;
            const float* w1 = w0 + N;
            #pragma unroll
            for (int i = 0; i < 4; ++i) {
                float4 a4 = *(const float4*)(w0 + i * 4);
                float4 b4 = *(const float4*)(w1 + i * 4);
                const float* fa = (const float*)&a4;
                const float* fb = (const float*)&b4;
                #pragma unroll
                for (int j = 0; j < 4; ++j)
                    *(unsigned int*)&Bs[(nl + i * 4 + j) * 40 + kp] = pack2(fa[j], fb[j]);
            }
        }
        __syncthreads();

        bf16x16 af[2];
        af[0] = load_frag_a(As, 40, wm * 32,      0);
        af[1] = load_frag_a(As, 40, wm * 32 + 16, 0);
        #pragma unroll
        for (int nt = 0; nt < 4; ++nt) {
            bf16x16 bfr = load_frag_b(Bs, 40, wn * 64 + nt * 16, 0);
            acc[0][nt] = wmma_bf16(af[0], bfr, acc[0][nt]);
            acc[1][nt] = wmma_bf16(af[1], bfr, acc[1][nt]);
        }
    }

    #pragma unroll
    for (int mt = 0; mt < 2; ++mt)
        #pragma unroll
        for (int nt = 0; nt < 4; ++nt)
            #pragma unroll
            for (int r = 0; r < 8; ++r) {
                int row = mb + wm * 32 + mt * 16 + r + 8 * half;
                int col = nb + wn * 64 + nt * 16 + (lane & 15);
                Y[(size_t)row * N + col] = f2bf(acc[mt][nt][r]);
            }
}

// ---------------------------------------------------------------------------
// Kernel 2: cor_embed bf16 [M, 64] = cor[M,2] @ Wcor[2,64]
// ---------------------------------------------------------------------------
__global__ void __launch_bounds__(256)
cor_embed_kernel(const float* __restrict__ cor, const float* __restrict__ Wcor,
                 unsigned short* __restrict__ ce) {
    int idx = blockIdx.x * 256 + threadIdx.x;          // M*64 threads
    int j = idx & 63, row = idx >> 6;
    float v = cor[row * 2] * Wcor[j] + cor[row * 2 + 1] * Wcor[64 + j];
    ce[idx] = f2bf(v);
}

// ---------------------------------------------------------------------------
// Kernel 3: attention. grid = B*NHEADS*2 blocks of 256 thr (8 waves).
// Block handles (b, h, half): 128 query rows; wave = 16 rows. LDS ~93KB.
// ---------------------------------------------------------------------------
__global__ void __launch_bounds__(256)
attn_kernel(const unsigned short* __restrict__ qb,
            const unsigned short* __restrict__ kvb,
            const unsigned short* __restrict__ ceb,
            const float* __restrict__ maskp,
            unsigned short* __restrict__ Xo,
            float* __restrict__ Crev) {
    __shared__ __align__(16) unsigned short VT[32 * 264];    // V^T  (d x m)
    __shared__ __align__(16) unsigned short CeT[16 * 264];   // Ce^T (j x m), rows 8..15 = 0
    __shared__ __align__(16) unsigned short Pb[128 * 264];   // softmax probs, bf16

    const int t    = threadIdx.x;
    const int wave = t >> 5;
    const int lane = t & 31;
    const int half = lane >> 4;
    const int bh   = blockIdx.x;             // 0 .. B*8*2-1
    const int b    = bh >> 4;
    const int h    = (bh >> 1) & 7;
    const int M0   = (bh & 1) * 128;
    const int qrow0 = M0 + wave * 16;

    // ---- stage V^T (waves 0-3) and Ce^T (waves 4-7): m-pairs, b32 stores
    if (t < 128) {
        const int m = t * 2;
        const unsigned short* vr0 = kvb + (size_t)(b * 256 + m) * 512 + 256 + h * 32;
        const unsigned short* vr1 = vr0 + 512;
        #pragma unroll
        for (int c = 0; c < 4; ++c) {
            union { uint4 q; unsigned short s[8]; } u0, u1;
            u0.q = *(const uint4*)(vr0 + c * 8);
            u1.q = *(const uint4*)(vr1 + c * 8);
            #pragma unroll
            for (int i = 0; i < 8; ++i)
                *(unsigned int*)&VT[(c * 8 + i) * 264 + m] =
                    (unsigned int)u0.s[i] | ((unsigned int)u1.s[i] << 16);
        }
    } else {
        const int m = (t - 128) * 2;
        const unsigned short* cr0 = ceb + (size_t)(b * 256 + m) * 64 + h * 8;
        const unsigned short* cr1 = cr0 + 64;
        union { uint4 q; unsigned short s[8]; } u0, u1;
        u0.q = *(const uint4*)(cr0);
        u1.q = *(const uint4*)(cr1);
        #pragma unroll
        for (int i = 0; i < 8; ++i) {
            *(unsigned int*)&CeT[i * 264 + m] =
                (unsigned int)u0.s[i] | ((unsigned int)u1.s[i] << 16);
            *(unsigned int*)&CeT[(i + 8) * 264 + m] = 0u;
        }
    }
    __syncthreads();

    // ---- S = Q K^T * scale + mask ---------------------------------------
    bf16x16 aq = load_frag_a(qb, 256, b * 256 + qrow0, h * 32);
    const float* mbase = maskp + (size_t)(b & (NWIN - 1)) * 256 * 256;

    float sc[16][8];
    #pragma unroll
    for (int nt = 0; nt < 16; ++nt) {
        bf16x16 kb = load_frag_b(kvb, 512, b * 256 + nt * 16, h * 32);
        f32x8 s = wmma_bf16(aq, kb, zero8());
        #pragma unroll
        for (int r = 0; r < 8; ++r) {
            int nrow = qrow0 + r + 8 * half;
            sc[nt][r] = s[r] * SCALE_F +
                        mbase[(size_t)nrow * 256 + nt * 16 + (lane & 15)];
        }
    }

    // ---- softmax (rows live in 16-lane half-groups) ---------------------
    float rinv[8], rmax[8];
    #pragma unroll
    for (int r = 0; r < 8; ++r) {
        float mx = sc[0][r];
        #pragma unroll
        for (int nt = 1; nt < 16; ++nt) mx = fmaxf(mx, sc[nt][r]);
        #pragma unroll
        for (int mk = 1; mk < 16; mk <<= 1) mx = fmaxf(mx, __shfl_xor(mx, mk, 32));
        rmax[r] = mx;
    }
    #pragma unroll
    for (int r = 0; r < 8; ++r) {
        float sum = 0.f;
        #pragma unroll
        for (int nt = 0; nt < 16; ++nt) {
            float e = __expf(sc[nt][r] - rmax[r]);
            sc[nt][r] = e;
            sum += e;
        }
        #pragma unroll
        for (int mk = 1; mk < 16; mk <<= 1) sum += __shfl_xor(sum, mk, 32);
        rinv[r] = 1.f / sum;
    }
    unsigned short* Pw = Pb + wave * 16 * 264;
    #pragma unroll
    for (int nt = 0; nt < 16; ++nt)
        #pragma unroll
        for (int r = 0; r < 8; ++r)
            Pw[(r + 8 * half) * 264 + nt * 16 + (lane & 15)] = f2bf(sc[nt][r] * rinv[r]);

    // ---- O = P V, Crev = P Ce -------------------------------------------
    f32x8 o0 = zero8(), o1 = zero8(), cc = zero8();
    #pragma unroll
    for (int kt = 0; kt < 8; ++kt) {
        bf16x16 ap = load_frag_a(Pw, 264, 0, kt * 32);
        bf16x16 b0 = load_frag_b(VT, 264, 0,  kt * 32);
        bf16x16 b1 = load_frag_b(VT, 264, 16, kt * 32);
        bf16x16 bc = load_frag_b(CeT, 264, 0, kt * 32);
        o0 = wmma_bf16(ap, b0, o0);
        o1 = wmma_bf16(ap, b1, o1);
        cc = wmma_bf16(ap, bc, cc);
    }
    #pragma unroll
    for (int r = 0; r < 8; ++r) {
        int grow = b * 256 + qrow0 + r + 8 * half;
        Xo[(size_t)grow * 256 + h * 32 + (lane & 15)]      = f2bf(o0[r]);
        Xo[(size_t)grow * 256 + h * 32 + 16 + (lane & 15)] = f2bf(o1[r]);
        if ((lane & 15) < 8)
            Crev[(size_t)grow * 64 + h * 8 + (lane & 15)] = cc[r];
    }
}

// ---------------------------------------------------------------------------
// Kernel 4: x_out f32 = X_bf16 @ Wproj + bproj.  Tile 64(M) x 256(N),
// grid 1024; A fragments come straight from global (already bf16).
// ---------------------------------------------------------------------------
__global__ void __launch_bounds__(256)
outproj_kernel(const unsigned short* __restrict__ Xb, const float* __restrict__ W,
               const float* __restrict__ bias, float* __restrict__ out) {
    __shared__ __align__(16) unsigned short Bs[256 * 40];

    const int t    = threadIdx.x;
    const int wave = t >> 5;
    const int lane = t & 31;
    const int half = lane >> 4;
    const int wm   = wave >> 2;     // 0..1
    const int wn   = wave & 3;      // 0..3
    const int mb   = blockIdx.x * 64;

    f32x8 acc[2][4];
    #pragma unroll
    for (int mt = 0; mt < 2; ++mt)
        #pragma unroll
        for (int nt = 0; nt < 4; ++nt) acc[mt][nt] = zero8();

    for (int ks = 0; ks < 8; ++ks) {
        __syncthreads();
        // stage W slice transposed: Bs[n][k] = W[ks*32+k][n], 32 x 256
        {
            const int kp = (t >> 4) * 2;
            const int nl = (t & 15) * 16;
            const float* w0 = W + (size_t)(ks * 32 + kp) * 256 + nl;
            const float* w1 = w0 + 256;
            #pragma unroll
            for (int i = 0; i < 4; ++i) {
                float4 a4 = *(const float4*)(w0 + i * 4);
                float4 b4 = *(const float4*)(w1 + i * 4);
                const float* fa = (const float*)&a4;
                const float* fb = (const float*)&b4;
                #pragma unroll
                for (int j = 0; j < 4; ++j)
                    *(unsigned int*)&Bs[(nl + i * 4 + j) * 40 + kp] = pack2(fa[j], fb[j]);
            }
        }
        __syncthreads();

        bf16x16 af[2];
        af[0] = load_frag_a(Xb, 256, mb + wm * 32,      ks * 32);
        af[1] = load_frag_a(Xb, 256, mb + wm * 32 + 16, ks * 32);
        #pragma unroll
        for (int nt = 0; nt < 4; ++nt) {
            bf16x16 bfr = load_frag_b(Bs, 40, wn * 64 + nt * 16, 0);
            acc[0][nt] = wmma_bf16(af[0], bfr, acc[0][nt]);
            acc[1][nt] = wmma_bf16(af[1], bfr, acc[1][nt]);
        }
    }

    #pragma unroll
    for (int mt = 0; mt < 2; ++mt)
        #pragma unroll
        for (int nt = 0; nt < 4; ++nt) {
            int col = wn * 64 + nt * 16 + (lane & 15);
            float bv = bias[col];
            #pragma unroll
            for (int r = 0; r < 8; ++r)
                out[(size_t)(mb + wm * 32 + mt * 16 + r + 8 * half) * 256 + col]
                    = acc[mt][nt][r] + bv;
        }
}

// ---------------------------------------------------------------------------
// Kernel 5: motion = (Crev - cor@Wcor) @ Wmotion + bmotion.  grid 512; 256 thr.
// ---------------------------------------------------------------------------
__global__ void __launch_bounds__(256)
motion_kernel(const float* __restrict__ Crev, const float* __restrict__ cor,
              const float* __restrict__ Wcor, const float* __restrict__ Wm,
              const float* __restrict__ bias, float* __restrict__ out) {
    __shared__ __align__(16) unsigned short WTm[64 * 72];   // WTm[n][k]

    const int t    = threadIdx.x;
    const int wave = t >> 5;
    const int lane = t & 31;
    const int half = lane >> 4;
    const int row0 = blockIdx.x * 128 + wave * 16;

    {   // stage transposed Wmotion with b32 stores of (k, k+1) pairs
        const int kp = (t >> 3) * 2;       // 0..62
        const int nl = (t & 7) * 8;        // 8 n cols
        const float* w0 = Wm + (size_t)kp * 64 + nl;
        const float* w1 = w0 + 64;
        #pragma unroll
        for (int i = 0; i < 2; ++i) {
            float4 a4 = *(const float4*)(w0 + i * 4);
            float4 b4 = *(const float4*)(w1 + i * 4);
            const float* fa = (const float*)&a4;
            const float* fb = (const float*)&b4;
            #pragma unroll
            for (int j = 0; j < 4; ++j)
                *(unsigned int*)&WTm[(nl + i * 4 + j) * 72 + kp] = pack2(fa[j], fb[j]);
        }
    }
    __syncthreads();

    const int m = lane & 15;
    const float c0 = cor[(size_t)(row0 + m) * 2];
    const float c1 = cor[(size_t)(row0 + m) * 2 + 1];

    f32x8 acc[4];
    #pragma unroll
    for (int nt = 0; nt < 4; ++nt) acc[nt] = zero8();

    #pragma unroll
    for (int kt = 0; kt < 2; ++kt) {
        // build A fragment on the fly: val = Crev - (c0*Wcor0 + c1*Wcor1)
        const int klo = kt * 32 + 8 * half;
        const float* cr = Crev + (size_t)(row0 + m) * 64 + klo;
        float4 q0 = *(const float4*)(cr);
        float4 q1 = *(const float4*)(cr + 4);
        float4 q2 = *(const float4*)(cr + 16);
        float4 q3 = *(const float4*)(cr + 20);
        float4 w00 = *(const float4*)(Wcor + klo);
        float4 w01 = *(const float4*)(Wcor + klo + 4);
        float4 w02 = *(const float4*)(Wcor + klo + 16);
        float4 w03 = *(const float4*)(Wcor + klo + 20);
        float4 w10 = *(const float4*)(Wcor + 64 + klo);
        float4 w11 = *(const float4*)(Wcor + 64 + klo + 4);
        float4 w12 = *(const float4*)(Wcor + 64 + klo + 16);
        float4 w13 = *(const float4*)(Wcor + 64 + klo + 20);
        const float* fq0 = (const float*)&q0; const float* fq1 = (const float*)&q1;
        const float* fq2 = (const float*)&q2; const float* fq3 = (const float*)&q3;
        const float* f00 = (const float*)&w00; const float* f01 = (const float*)&w01;
        const float* f02 = (const float*)&w02; const float* f03 = (const float*)&w03;
        const float* f10 = (const float*)&w10; const float* f11 = (const float*)&w11;
        const float* f12 = (const float*)&w12; const float* f13 = (const float*)&w13;
        float vlo[8], vhi[8];
        #pragma unroll
        for (int i = 0; i < 4; ++i) {
            vlo[i]     = fq0[i] - (c0 * f00[i] + c1 * f10[i]);
            vlo[i + 4] = fq1[i] - (c0 * f01[i] + c1 * f11[i]);
            vhi[i]     = fq2[i] - (c0 * f02[i] + c1 * f12[i]);
            vhi[i + 4] = fq3[i] - (c0 * f03[i] + c1 * f13[i]);
        }
        B16Frag af;
        #pragma unroll
        for (int i = 0; i < 4; ++i) {
            af.u[i]     = pack2(vlo[2 * i], vlo[2 * i + 1]);
            af.u[i + 4] = pack2(vhi[2 * i], vhi[2 * i + 1]);
        }
        #pragma unroll
        for (int nt = 0; nt < 4; ++nt) {
            bf16x16 bfr = load_frag_b(WTm, 72, nt * 16, kt * 32);
            acc[nt] = wmma_bf16(af.v, bfr, acc[nt]);
        }
    }
    #pragma unroll
    for (int nt = 0; nt < 4; ++nt) {
        int col = nt * 16 + (lane & 15);
        float bv = bias[col];
        #pragma unroll
        for (int r = 0; r < 8; ++r)
            out[(size_t)(row0 + r + 8 * half) * 64 + col] = acc[nt][r] + bv;
    }
}

// ---------------------------------------------------------------------------
extern "C" void kernel_launch(void* const* d_in, const int* in_sizes, int n_in,
                              void* d_out, int out_size, void* d_ws, size_t ws_size,
                              hipStream_t stream) {
    const float* x1    = (const float*)d_in[0];
    const float* x2    = (const float*)d_in[1];
    const float* cor   = (const float*)d_in[2];
    const float* maskp = (const float*)d_in[3];
    const float* Wq    = (const float*)d_in[4];
    const float* Wkv   = (const float*)d_in[5];
    const float* Wcor  = (const float*)d_in[6];
    const float* Wproj = (const float*)d_in[7];
    const float* bproj = (const float*)d_in[8];
    const float* Wmot  = (const float*)d_in[9];
    const float* bmot  = (const float*)d_in[10];

    char* wsb = (char*)d_ws;
    unsigned short* qb   = (unsigned short*)(wsb);                  // 32 MB
    unsigned short* kvb  = (unsigned short*)(wsb + 33554432);       // 64 MB
    unsigned short* ceb  = (unsigned short*)(wsb + 100663296);      //  8 MB
    unsigned short* Xb   = (unsigned short*)(wsb + 109051904);      // 32 MB
    float*          Crev = (float*)        (wsb + 142606336);       // 16 MB

    float* xout = (float*)d_out;
    float* mout = xout + (size_t)MROWS * DIM_;

    gemm_f32_bf16<<<dim3(1024, 1), 256, 0, stream>>>(x1, Wq,  qb,  256, 256);
    gemm_f32_bf16<<<dim3(1024, 2), 256, 0, stream>>>(x2, Wkv, kvb, 256, 512);
    cor_embed_kernel<<<(MROWS * 64) / 256, 256, 0, stream>>>(cor, Wcor, ceb);
    attn_kernel<<<BATCH * NHEADS * 2, 256, 0, stream>>>(qb, kvb, ceb, maskp, Xb, Crev);
    outproj_kernel<<<1024, 256, 0, stream>>>(Xb, Wproj, bproj, xout);
    motion_kernel<<<512, 256, 0, stream>>>(Crev, cor, Wcor, Wmot, bmot, mout);
}
